// BlockSparseThresLinear_82798379532751
// MI455X (gfx1250) — matrix operand order repaired and checked by reading, not compiled, not run on hardware
//
#include <hip/hip_runtime.h>

// Problem constants (from reference)
#define BSZ   8192
#define D_IN  4096
#define D_OUT 4096
#define NM    128          // BSZ/64
#define NK    64           // D_IN/64
#define THRES_SUM 3276.8f  // 0.8 * 64 * 64

typedef __attribute__((ext_vector_type(16))) __bf16 v16bf;
typedef __attribute__((ext_vector_type(8)))  __bf16 v8bf;
typedef __attribute__((ext_vector_type(4)))  __bf16 v4bf;
typedef __attribute__((ext_vector_type(8)))  float  v8f;

// Workgroup barrier that orders ONLY LDS traffic (DScnt), leaving in-flight
// global loads (LOADcnt) running underneath the WMMA burst. __syncthreads()
// would drain LOADcnt too (s_wait_loadcnt_dscnt 0x0) and kill the pipeline.
__device__ __forceinline__ void wg_barrier_lds_only() {
    asm volatile(
        "s_wait_dscnt 0x0\n\t"
        "s_barrier_signal -1\n\t"
        "s_barrier_wait -1"
        ::: "memory");
}

// ---------------------------------------------------------------------------
// Kernel 1: per-64x64-block mean-|x| mask -> int mask[NM*NK] in d_ws
// ---------------------------------------------------------------------------
__global__ __launch_bounds__(256) void bs_mask_kernel(const float* __restrict__ x,
                                                      int* __restrict__ mask) {
    __shared__ float red[256];
    const int tid = threadIdx.x;
    const int bid = blockIdx.x;       // [0, NM*NK)
    const int bm  = bid >> 6;         // /NK
    const int bk  = bid & 63;
    const float* base = x + (size_t)bm * 64 * D_IN + (size_t)bk * 64;
    float s = 0.f;
#pragma unroll
    for (int i = 0; i < 4; ++i) {
        const int f   = tid + i * 256;     // 1024 float4 per block
        const int row = f >> 4;
        const int c4  = f & 15;
        const float4 v = *(const float4*)(base + (size_t)row * D_IN + c4 * 4);
        s += fabsf(v.x) + fabsf(v.y) + fabsf(v.z) + fabsf(v.w);
    }
    red[tid] = s;
    __syncthreads();
    for (int off = 128; off > 0; off >>= 1) {
        if (tid < off) red[tid] += red[tid + off];
        __syncthreads();
    }
    if (tid == 0) mask[bid] = (red[0] > THRES_SUM) ? 1 : 0;
}

// ---------------------------------------------------------------------------
// Kernel 2: block-sparse GEMM, bf16x3 split precision, double-buffered LDS
// software pipeline: issue global loads for the NEXT active k-block, run the
// 24 WMMAs of the current buffer while they are in flight, then cvt+store.
// Tile: 64(M) x 128(N); grid n-fast / mb-slow keeps w (67MB) in the 192MB L2.
// ---------------------------------------------------------------------------
#define LDK 72   // padded bf16 K-stride: 144 bytes, multiple of 16 -> aligned b128

__global__ __launch_bounds__(256) void bs_gemm_kernel(const float* __restrict__ x,
                                                      const float* __restrict__ w,
                                                      const int*   __restrict__ mask,
                                                      float*       __restrict__ out) {
    __shared__ __bf16 As_hi[2][64][LDK];
    __shared__ __bf16 As_lo[2][64][LDK];
    __shared__ __bf16 Bs_hi[2][128][LDK];   // transposed: [n][k]
    __shared__ __bf16 Bs_lo[2][128][LDK];

    const int tid  = threadIdx.x;
    const int lane = tid & 31;
    const int wv   = tid >> 5;           // wave id 0..7
    const int ln   = lane & 15;
    const int lh   = lane >> 4;          // lane half
    const int mb   = blockIdx.y;         // 64-row block == mask row (slow dim)
    const int gm0  = mb * 64;
    const int gn0  = blockIdx.x * 128;   // fast dim: consecutive WGs share x tile

    // w-staging mapping: thread owns column n and half of the k range
    const int wn  = tid & 127;
    const int wkh = tid >> 7;

    // staged-but-unconverted data held in registers across the WMMA burst
    float4 xr[4];
    float  wr[32];

    auto load_issue = [&](int kb) {
        const float* xb = x + (size_t)gm0 * D_IN + (size_t)kb * 64;
#pragma unroll
        for (int i = 0; i < 4; ++i) {
            const int f   = tid + i * 256;
            const int row = f >> 4;
            const int c4  = f & 15;
            xr[i] = *(const float4*)(xb + (size_t)row * D_IN + c4 * 4);
        }
        const float* wb = w + (size_t)kb * 64 * D_OUT + gn0 + wn;  // coalesced b32
#pragma unroll
        for (int j = 0; j < 8; ++j) {
            const int k0 = wkh * 32 + j * 4;
#pragma unroll
            for (int jj = 0; jj < 4; ++jj)
                wr[j * 4 + jj] = wb[(size_t)(k0 + jj) * D_OUT];
        }
    };

    auto convert_store = [&](int buf) {
#pragma unroll
        for (int i = 0; i < 4; ++i) {
            const int f   = tid + i * 256;
            const int row = f >> 4;
            const int c4  = f & 15;
            const float vv[4] = {xr[i].x, xr[i].y, xr[i].z, xr[i].w};
            v4bf hi, lo;
#pragma unroll
            for (int j = 0; j < 4; ++j) {
                const __bf16 h = (__bf16)vv[j];
                hi[j] = h;
                lo[j] = (__bf16)(vv[j] - (float)h);
            }
            *(v4bf*)&As_hi[buf][row][c4 * 4] = hi;
            *(v4bf*)&As_lo[buf][row][c4 * 4] = lo;
        }
#pragma unroll
        for (int j = 0; j < 8; ++j) {
            const int k0 = wkh * 32 + j * 4;
            v4bf hi, lo;
#pragma unroll
            for (int jj = 0; jj < 4; ++jj) {
                const float v = wr[j * 4 + jj];
                const __bf16 h = (__bf16)v;
                hi[jj] = h;
                lo[jj] = (__bf16)(v - (float)h);
            }
            *(v4bf*)&Bs_hi[buf][wn][k0] = hi;   // contiguous-k b64 stores
            *(v4bf*)&Bs_lo[buf][wn][k0] = lo;
        }
    };

    v8f acc[4];
#pragma unroll
    for (int mt = 0; mt < 4; ++mt)
        acc[mt] = (v8f){0.f, 0.f, 0.f, 0.f, 0.f, 0.f, 0.f, 0.f};

    auto compute_buf = [&](int buf) {
#pragma unroll
        for (int ko = 0; ko < 64; ko += 32) {
            // B fragment: lane -> column n; lanes 0-15 K 0..7 & 16..23, 16-31 +8
            const int bn = wv * 16 + ln;
            const int bk = ko + lh * 8;
            const v8bf b0h = *(const v8bf*)&Bs_hi[buf][bn][bk];
            const v8bf b1h = *(const v8bf*)&Bs_hi[buf][bn][bk + 16];
            const v8bf b0l = *(const v8bf*)&Bs_lo[buf][bn][bk];
            const v8bf b1l = *(const v8bf*)&Bs_lo[buf][bn][bk + 16];
            const v16bf bh = __builtin_shufflevector(b0h, b1h, 0,1,2,3,4,5,6,7,8,9,10,11,12,13,14,15);
            const v16bf bl = __builtin_shufflevector(b0l, b1l, 0,1,2,3,4,5,6,7,8,9,10,11,12,13,14,15);
#pragma unroll
            for (int mt = 0; mt < 4; ++mt) {
                // A fragment: lane -> row m; lanes 0-15 K 0..7 & 16..23, 16-31 +8
                const int ar = mt * 16 + ln;
                const int ak = ko + lh * 8;
                const v8bf a0h = *(const v8bf*)&As_hi[buf][ar][ak];
                const v8bf a1h = *(const v8bf*)&As_hi[buf][ar][ak + 16];
                const v8bf a0l = *(const v8bf*)&As_lo[buf][ar][ak];
                const v8bf a1l = *(const v8bf*)&As_lo[buf][ar][ak + 16];
                const v16bf ah = __builtin_shufflevector(a0h, a1h, 0,1,2,3,4,5,6,7,8,9,10,11,12,13,14,15);
                const v16bf al = __builtin_shufflevector(a0l, a1l, 0,1,2,3,4,5,6,7,8,9,10,11,12,13,14,15);
                // bf16x3: hi*hi + lo*hi + hi*lo (lo*lo negligible)
                acc[mt] = __builtin_amdgcn_wmma_f32_16x16x32_bf16(
                    false, ah, false, bh, (short)0, acc[mt], false, false);
                acc[mt] = __builtin_amdgcn_wmma_f32_16x16x32_bf16(
                    false, al, false, bh, (short)0, acc[mt], false, false);
                acc[mt] = __builtin_amdgcn_wmma_f32_16x16x32_bf16(
                    false, ah, false, bl, (short)0, acc[mt], false, false);
            }
        }
    };

    // ---- block-sparse pipelined main loop (mask scan is uniform/scalar) ----
    const int mrow = mb * NK;
    int kb = 0;
    while (kb < NK && !mask[mrow + kb]) ++kb;
    if (kb < NK) {
        load_issue(kb);          // prologue: fully stage first active block
        convert_store(0);
        int cur = 0;
        for (;;) {
            int kn = kb + 1;     // scalar look-ahead for next active k-block
            while (kn < NK && !mask[mrow + kn]) ++kn;
            const bool have_next = (kn < NK);
            if (have_next) load_issue(kn);   // global loads fly under the WMMAs
            wg_barrier_lds_only();           // DScnt-only barrier: LOADcnt untouched
            compute_buf(cur);
            if (!have_next) break;
            convert_store(cur ^ 1);          // s_wait_loadcnt here, cvt, b64 ds_stores
            cur ^= 1;
            kb = kn;
        }
    }

    // ---- store: C/D layout — VGPR r, lane -> (m = r + 8*lh, n = ln) ----
    float* ob = out + (size_t)gm0 * D_OUT + gn0 + wv * 16 + ln;
#pragma unroll
    for (int mt = 0; mt < 4; ++mt) {
#pragma unroll
        for (int r = 0; r < 8; ++r) {
            ob[(size_t)(mt * 16 + r + 8 * lh) * D_OUT] = acc[mt][r];
        }
    }
}

// ---------------------------------------------------------------------------
extern "C" void kernel_launch(void* const* d_in, const int* in_sizes, int n_in,
                              void* d_out, int out_size, void* d_ws, size_t ws_size,
                              hipStream_t stream) {
    const float* x = (const float*)d_in[0];
    const float* w = (const float*)d_in[1];
    int*   mask = (int*)d_ws;            // NM*NK ints = 32 KB
    float* out  = (float*)d_out;

    bs_mask_kernel<<<NM * NK, 256, 0, stream>>>(x, mask);

    dim3 grid(D_OUT / 128, NM);          // n fast (32), mb slow (128): w stays in L2
    bs_gemm_kernel<<<grid, 256, 0, stream>>>(x, w, mask, out);
}
